// DenseGSSD_22969485099258
// MI455X (gfx1250) — matrix operand order repaired and checked by reading, not compiled, not run on hardware
//
#include <hip/hip_runtime.h>
#include <cstdint>

typedef _Float16 v16h __attribute__((ext_vector_type(16)));
typedef float v8f __attribute__((ext_vector_type(8)));

namespace {

constexpr int kDim = 192;
constexpr int kDInner = 384;
constexpr int kNH = 48;
constexpr int kHD = 8;
constexpr int kDState = 64;
constexpr int kConvDim = 512;    // D_INNER + 2*D_STATE
constexpr int kDProj = 944;      // 2*384 + 2*64 + 48
constexpr int kChunk = 64;
constexpr int kSeq = 4096;
constexpr int kBatch = 4;
constexpr int kNChunks = kSeq / kChunk;  // 64
constexpr int kRows = kBatch * kSeq;     // 16384

// --- WMMA fragment index maps (CDNA5 ISA 7.12.2, 16-bit operands, wave32) ---
__device__ __forceinline__ int a_k(int j, int g) { return j + ((j >= 8) ? 8 : 0) + 8 * g; }
__device__ __forceinline__ int b_k(int j, int g) { return j + ((j >= 8) ? 8 : 0) + 8 * g; }

__device__ __forceinline__ float wave_sum(float v) {
#pragma unroll
  for (int m = 16; m >= 1; m >>= 1) v += __shfl_xor(v, m, 32);
  return v;
}

__device__ __forceinline__ float sigmoidf_(float x) { return 1.0f / (1.0f + __expf(-x)); }
__device__ __forceinline__ float siluf_(float x) { return x * sigmoidf_(x); }
__device__ __forceinline__ float softplusf_(float x) {
  return (x > 20.0f) ? x : log1pf(__expf(x));
}

// ---- CDNA5 async global->LDS copy (ASYNCcnt path, ISA 08_async_tensor) -----
__device__ __forceinline__ void async_copy_b128(uint32_t lds_off, const float* gptr) {
  asm volatile("global_load_async_to_lds_b128 %0, %1, off"
               :
               : "v"(lds_off), "v"(gptr)
               : "memory");
}
__device__ __forceinline__ void wait_async0() {
  asm volatile("s_wait_asynccnt 0" ::: "memory");
}

// ---------------- LayerNorm over 192 cols, one wave per row ----------------
__global__ void ln_rows_k(const float* __restrict__ in, const float* __restrict__ w,
                          const float* __restrict__ b, float* __restrict__ out, float eps) {
  int lane = threadIdx.x & 31;
  int row = blockIdx.x * 8 + (threadIdx.x >> 5);
  const float* ip = in + (size_t)row * kDim;
  float v[6], s = 0.f;
#pragma unroll
  for (int i = 0; i < 6; ++i) { v[i] = ip[lane + 32 * i]; s += v[i]; }
  float mean = wave_sum(s) * (1.0f / kDim);
  float q = 0.f;
#pragma unroll
  for (int i = 0; i < 6; ++i) { float d = v[i] - mean; q += d * d; }
  float var = wave_sum(q) * (1.0f / kDim);
  float r = rsqrtf(var + eps);
  float* op = out + (size_t)row * kDim;
#pragma unroll
  for (int i = 0; i < 6; ++i) {
    int c = lane + 32 * i;
    op[c] = (v[i] - mean) * r * w[c] + b[c];
  }
}

// ------------- C = A(MxK, f32 row-major) * W(NxK, f32 row-major)^T ----------
// 256-thread block = 8 waves computing a 128x16 macro-tile. Per K-step the
// 128x32 A tile and 16x32 B tile are staged into LDS via async b128 copies,
// double-buffered so next-stage loads overlap current WMMAs. B tile is shared
// by all 8 waves. Requires M%128==0, N%16==0, K%32==0.
__global__ void gemm_wmma_k(const float* __restrict__ A, int lda,
                            const float* __restrict__ W, int ldw,
                            float* __restrict__ C, int ldc, int K) {
  __shared__ __align__(16) float sA[2][128 * 32];  // 2 x 16 KB
  __shared__ __align__(16) float sB[2][16 * 32];   // 2 x 2 KB
  int tid = threadIdx.x, lane = tid & 31, wid = tid >> 5;
  int g = lane >> 4, q = lane & 15;
  int n0 = blockIdx.x * 16;
  int m0 = blockIdx.y * 128;
  uint32_t aBase = (uint32_t)(uintptr_t)(void*)&sA[0][0];
  uint32_t bBase = (uint32_t)(uintptr_t)(void*)&sB[0][0];
  const int KT = K / 32;

  auto stage = [&](int kt, int buf) {
    const float* Ag = A + (size_t)m0 * lda + kt * 32;
#pragma unroll
    for (int i = 0; i < 4; ++i) {
      int idx = tid + i * 256;  // 1024 x 16B chunks = 128x32 f32
      int r = idx >> 3, cg = (idx & 7) * 4;
      async_copy_b128(aBase + (uint32_t)(buf * 128 * 32 + r * 32 + cg) * 4u,
                      Ag + (size_t)r * lda + cg);
    }
    if (tid < 128) {  // 128 x 16B chunks = 16x32 f32
      int r = tid >> 3, cg = (tid & 7) * 4;
      async_copy_b128(bBase + (uint32_t)(buf * 16 * 32 + r * 32 + cg) * 4u,
                      W + (size_t)(n0 + r) * ldw + kt * 32 + cg);
    }
  };

  v8f acc = {};
  stage(0, 0);
  for (int kt = 0; kt < KT; ++kt) {
    int buf = kt & 1;
    wait_async0();
    __syncthreads();  // staged tile visible to all waves
    if (kt + 1 < KT) stage(kt + 1, buf ^ 1);
    const float* a0 = &sA[buf][(wid * 16 + q) * 32];
    const float* b0 = &sB[buf][q * 32];
    v16h a, bm;
#pragma unroll
    for (int j = 0; j < 16; ++j) {
      a[j] = (_Float16)a0[a_k(j, g)];
      bm[j] = (_Float16)b0[b_k(j, g)];
    }
    acc = __builtin_amdgcn_wmma_f32_16x16x32_f16(false, a, false, bm, (short)0, acc, false, false);
    __syncthreads();  // done reading buf before it is restaged
  }
  float* Cp = C + (size_t)(m0 + wid * 16) * ldc + n0 + q;
#pragma unroll
  for (int r = 0; r < 8; ++r) Cp[(size_t)(r + 8 * g) * ldc] = acc[r];
}

// ---------------- dt = softplus(zx[:, -48:] + dt_bias) ---------------------
__global__ void dt_k(const float* __restrict__ zx, const float* __restrict__ dt_bias,
                     float* __restrict__ dt) {
  int idx = blockIdx.x * 256 + threadIdx.x;  // kRows * kNH threads, exact
  int row = idx / kNH, h = idx % kNH;
  dt[idx] = softplusf_(zx[(size_t)row * kDProj + (kDProj - kNH) + h] + dt_bias[h]);
}

// --------- depthwise causal conv (D_CONV=4) over seq + bias + SiLU ----------
__global__ void conv1d_k(const float* __restrict__ zx, const float* __restrict__ cw,
                         const float* __restrict__ cb, float* __restrict__ xc) {
  int idx = blockIdx.x * 256 + threadIdx.x;  // kRows * 512 threads, exact
  int row = idx >> 9, ch = idx & 511;
  int l = row & (kSeq - 1);
  float acc = cb[ch];
#pragma unroll
  for (int k = 0; k < 4; ++k) {
    int lp = l + k - 3;
    if (lp >= 0) acc += zx[(size_t)(row + k - 3) * kDProj + kDInner + ch] * cw[ch * 4 + k];
  }
  xc[idx] = siluf_(acc);
}

// ----------- xdt = x*dt (per head broadcast), dA = dt * (-exp(A_log)) -------
__global__ void xdt_k(const float* __restrict__ xc, const float* __restrict__ dt,
                      const float* __restrict__ A_log, float* __restrict__ xdt,
                      float* __restrict__ dA) {
  int idx = blockIdx.x * 256 + threadIdx.x;  // kRows * kNH
  int row = idx / kNH, h = idx % kNH;
  float d = dt[idx];
  dA[idx] = -d * __expf(A_log[h]);
#pragma unroll
  for (int p = 0; p < kHD; ++p)
    xdt[(size_t)row * kDInner + h * kHD + p] = xc[(size_t)row * kConvDim + h * kHD + p] * d;
}

// ---------------- chunk-local SSD: CB, Y_diag, states, A_cs -----------------
// Grid (chunk, batch), 256 threads (8 waves). All GEMMs are 16x16x32 f16 WMMA.
__global__ void ssd_chunk_k(const float* __restrict__ dA, const float* __restrict__ xc,
                            const float* __restrict__ xdt, float* __restrict__ Y,
                            float* __restrict__ states, float* __restrict__ cdec,
                            float* __restrict__ Acs) {
  __shared__ _Float16 sB[kChunk * kDState];   // 8 KB
  __shared__ _Float16 sC[kChunk * kDState];   // 8 KB
  __shared__ float sAcs[kNH * kChunk];        // 12 KB
  __shared__ float sCB[kChunk * kChunk];      // 16 KB
  int tid = threadIdx.x, lane = tid & 31, wid = tid >> 5;
  int g = lane >> 4, q = lane & 15;
  int cc = blockIdx.x, bb = blockIdx.y;
  size_t rowBase = (size_t)(bb * kSeq + cc * kChunk);

  if (tid < kNH) {  // serial cumsum of dA per head (64 steps)
    float a = 0.f;
    for (int s = 0; s < kChunk; ++s) {
      a += dA[(rowBase + s) * kNH + tid];
      sAcs[tid * kChunk + s] = a;
    }
  }
  for (int i = tid; i < kChunk * kDState; i += 256) {
    int l = i >> 6, n = i & 63;
    sB[i] = (_Float16)xc[(rowBase + l) * kConvDim + kDInner + n];
    sC[i] = (_Float16)xc[(rowBase + l) * kConvDim + kDInner + kDState + n];
  }
  __syncthreads();

  // CB[l][s] = sum_n C[l][n]*B[s][n]  (64x64x64) -> 16 tiles, 2 per wave
#pragma unroll
  for (int t = 0; t < 2; ++t) {
    int tile = wid * 2 + t;
    int m0 = (tile >> 2) * 16, n0 = (tile & 3) * 16;
    v8f acc = {};
    for (int k0 = 0; k0 < kDState; k0 += 32) {
      v16h a, bm;
#pragma unroll
      for (int j = 0; j < 16; ++j) {
        a[j] = sC[(m0 + q) * kDState + k0 + a_k(j, g)];
        bm[j] = sB[(n0 + q) * kDState + k0 + b_k(j, g)];
      }
      acc = __builtin_amdgcn_wmma_f32_16x16x32_f16(false, a, false, bm, (short)0, acc, false, false);
    }
#pragma unroll
    for (int r = 0; r < 8; ++r) sCB[(m0 + r + 8 * g) * kChunk + n0 + q] = acc[r];
  }
  __syncthreads();

  for (int h = wid; h < kNH; h += 8) {
    const float* Ah = sAcs + h * kChunk;
    const float* Xp = xdt + rowBase * kDInner + h * kHD;  // [s*384 + p]

    // Y_diag[l][p] = sum_s (mask*exp(Acs[l]-Acs[s])*CB[l][s]) * X[s][p]
#pragma unroll
    for (int mt = 0; mt < 4; ++mt) {
      int m0 = mt * 16;
      int l = m0 + q;
      v8f acc = {};
      for (int k0 = 0; k0 < kChunk; k0 += 32) {
        v16h a, bm;
#pragma unroll
        for (int j = 0; j < 16; ++j) {
          int s = k0 + a_k(j, g);
          float lm = (s <= l) ? __expf(Ah[l] - Ah[s]) * sCB[l * kChunk + s] : 0.f;
          a[j] = (_Float16)lm;
          int s2 = k0 + b_k(j, g);
          bm[j] = (q < kHD) ? (_Float16)Xp[(size_t)s2 * kDInner + q] : (_Float16)0.f;
        }
        acc = __builtin_amdgcn_wmma_f32_16x16x32_f16(false, a, false, bm, (short)0, acc, false, false);
      }
      if (q < kHD) {
#pragma unroll
        for (int r = 0; r < 8; ++r) {
          int lr = m0 + r + 8 * g;
          Y[(rowBase + lr) * kDInner + h * kHD + q] = acc[r];
        }
      }
    }

    // states[p][n] = sum_s X[s][p]*exp(Acs[63]-Acs[s]) * B[s][n]  (p padded to 16)
    float Alast = Ah[kChunk - 1];
#pragma unroll
    for (int nt = 0; nt < 4; ++nt) {
      int n0 = nt * 16;
      v8f acc = {};
      for (int k0 = 0; k0 < kChunk; k0 += 32) {
        v16h a, bm;
#pragma unroll
        for (int j = 0; j < 16; ++j) {
          int s = k0 + a_k(j, g);
          float av = (q < kHD) ? (float)Xp[(size_t)s * kDInner + q] * __expf(Alast - Ah[s]) : 0.f;
          a[j] = (_Float16)av;
          int s2 = k0 + b_k(j, g);
          bm[j] = sB[s2 * kDState + n0 + q];
        }
        acc = __builtin_amdgcn_wmma_f32_16x16x32_f16(false, a, false, bm, (short)0, acc, false, false);
      }
      if (g == 0) {  // valid rows p = 0..7 live in lanes 0-15, regs r=0..7
#pragma unroll
        for (int r = 0; r < 8; ++r)
          states[((size_t)((bb * kNChunks + cc) * kNH + h) * kHD + r) * kDState + n0 + q] = acc[r];
      }
    }

    for (int s = lane; s < kChunk; s += 32)
      Acs[((size_t)(bb * kNChunks + cc) * kNH + h) * kChunk + s] = Ah[s];
    if (lane == 0) cdec[(size_t)(bb * kNChunks + cc) * kNH + h] = __expf(Ah[kChunk - 1]);
  }
}

// ---------------- sequential inter-chunk scan (per b,h,p,n) -----------------
__global__ void scan_k(const float* __restrict__ states, const float* __restrict__ cdec,
                       float* __restrict__ prev) {
  int idx = blockIdx.x * 256 + threadIdx.x;  // kBatch*kNH*kHD*kDState = 98304
  int n = idx & 63, p = (idx >> 6) & 7;
  int h = (idx >> 9) % kNH;
  int b = (idx >> 9) / kNH;
  float carry = 0.f;
  for (int c = 0; c < kNChunks; ++c) {
    size_t base = ((size_t)((b * kNChunks + c) * kNH + h) * kHD + p) * kDState + n;
    prev[base] = carry;
    carry = carry * cdec[(size_t)(b * kNChunks + c) * kNH + h] + states[base];
  }
}

// ------- Y += exp(Acs[l]) * (C[l][n] . prev[h][p][n])   (per b,chunk) -------
__global__ void ssd_yoff_k(const float* __restrict__ xc, const float* __restrict__ Acs,
                           const float* __restrict__ prev, float* __restrict__ Y) {
  __shared__ _Float16 sC[kChunk * kDState];  // 8 KB
  __shared__ float sAcs[kNH * kChunk];       // 12 KB
  int tid = threadIdx.x, lane = tid & 31, wid = tid >> 5;
  int g = lane >> 4, q = lane & 15;
  int cc = blockIdx.x, bb = blockIdx.y;
  size_t rowBase = (size_t)(bb * kSeq + cc * kChunk);
  for (int i = tid; i < kChunk * kDState; i += 256) {
    int l = i >> 6, n = i & 63;
    sC[i] = (_Float16)xc[(rowBase + l) * kConvDim + kDInner + kDState + n];
  }
  for (int i = tid; i < kNH * kChunk; i += 256)
    sAcs[i] = Acs[(size_t)(bb * kNChunks + cc) * kNH * kChunk + i];
  __syncthreads();

  for (int h = wid; h < kNH; h += 8) {
    const float* Pv = prev + (size_t)((bb * kNChunks + cc) * kNH + h) * kHD * kDState;
#pragma unroll
    for (int mt = 0; mt < 4; ++mt) {
      int m0 = mt * 16;
      v8f acc = {};
      for (int k0 = 0; k0 < kDState; k0 += 32) {
        v16h a, bm;
#pragma unroll
        for (int j = 0; j < 16; ++j) {
          a[j] = sC[(m0 + q) * kDState + k0 + a_k(j, g)];
          int n = k0 + b_k(j, g);
          bm[j] = (q < kHD) ? (_Float16)Pv[(size_t)q * kDState + n] : (_Float16)0.f;
        }
        acc = __builtin_amdgcn_wmma_f32_16x16x32_f16(false, a, false, bm, (short)0, acc, false, false);
      }
      if (q < kHD) {
#pragma unroll
        for (int r = 0; r < 8; ++r) {
          int lr = m0 + r + 8 * g;
          size_t o = (rowBase + lr) * kDInner + h * kHD + q;
          Y[o] += __expf(sAcs[h * kChunk + lr]) * acc[r];
        }
      }
    }
  }
}

// -------- y = (Y + x*D[h]) * silu(z); RMSNorm over 384 with norm_w ----------
__global__ void gate_rms_k(const float* __restrict__ Y, const float* __restrict__ xc,
                           const float* __restrict__ zx, const float* __restrict__ Dp,
                           const float* __restrict__ nw, float* __restrict__ out) {
  int lane = threadIdx.x & 31;
  int row = blockIdx.x * 8 + (threadIdx.x >> 5);
  float v[12], ss = 0.f;
#pragma unroll
  for (int i = 0; i < 12; ++i) {
    int c = lane + 32 * i;
    float y = Y[(size_t)row * kDInner + c] + xc[(size_t)row * kConvDim + c] * Dp[c >> 3];
    float z = zx[(size_t)row * kDProj + c];
    float t = y * siluf_(z);
    v[i] = t;
    ss += t * t;
  }
  float ms = wave_sum(ss) * (1.0f / kDInner);
  float r = rsqrtf(ms + 1e-5f);
#pragma unroll
  for (int i = 0; i < 12; ++i) {
    int c = lane + 32 * i;
    out[(size_t)row * kDInner + c] = v[i] * r * nw[c];
  }
}

// ----- x1 = u*ss1 + m; out = x1*ss2 + LN(x1) + optional residuals -----------
__global__ void finalize_k(const float* __restrict__ u, const float* __restrict__ m,
                           const float* __restrict__ ss1, const float* __restrict__ ss2,
                           const float* __restrict__ w, const float* __restrict__ b,
                           const float* __restrict__ r1, const float* __restrict__ r2,
                           const float* __restrict__ r3, float* __restrict__ out) {
  int lane = threadIdx.x & 31;
  int row = blockIdx.x * 8 + (threadIdx.x >> 5);
  float x1[6], s = 0.f;
#pragma unroll
  for (int i = 0; i < 6; ++i) {
    int c = lane + 32 * i;
    x1[i] = u[(size_t)row * kDim + c] * ss1[c] + m[(size_t)row * kDim + c];
    s += x1[i];
  }
  float mean = wave_sum(s) * (1.0f / kDim);
  float qv = 0.f;
#pragma unroll
  for (int i = 0; i < 6; ++i) { float d = x1[i] - mean; qv += d * d; }
  float var = wave_sum(qv) * (1.0f / kDim);
  float rr = rsqrtf(var + 1e-5f);
#pragma unroll
  for (int i = 0; i < 6; ++i) {
    int c = lane + 32 * i;
    size_t o = (size_t)row * kDim + c;
    float v = x1[i] * ss2[c] + (x1[i] - mean) * rr * w[c] + b[c];
    if (r1) v += r1[o];
    if (r2) v += r2[o];
    if (r3) v += r3[o];
    out[o] = v;
  }
}

// --------- 3x3 SAME conv as implicit GEMM: M=16384 px, N=192, K=1728 --------
__global__ void conv2d_wmma_k(const float* __restrict__ X, const float* __restrict__ W,
                              const float* __restrict__ bias, float* __restrict__ out) {
  int lane = threadIdx.x & 31, g = lane >> 4, q = lane & 15;
  int n0 = blockIdx.x * 16, m0 = blockIdx.y * 16;
  int mm = m0 + q;
  int bimg = mm >> 12, l = mm & 4095, yy = l >> 6, xx = l & 63;
  v8f acc = {};
  for (int k0 = 0; k0 < 9 * kDim; k0 += 32) {
    v16h a, bm;
#pragma unroll
    for (int j = 0; j < 16; ++j) {
      {
        int kk = k0 + a_k(j, g);
        int tap = kk / kDim, cin = kk - tap * kDim;
        int py = yy + tap / 3 - 1, px = xx + tap % 3 - 1;
        float v = 0.f;
        if (py >= 0 && py < 64 && px >= 0 && px < 64)
          v = X[((size_t)(bimg << 12) + (py << 6) + px) * kDim + cin];
        a[j] = (_Float16)v;
      }
      {
        int kk = k0 + b_k(j, g);
        int tap = kk / kDim, cin = kk - tap * kDim;
        bm[j] = (_Float16)W[(((size_t)(n0 + q) * kDim + cin) * 3 + tap / 3) * 3 + (tap % 3)];
      }
    }
    acc = __builtin_amdgcn_wmma_f32_16x16x32_f16(false, a, false, bm, (short)0, acc, false, false);
  }
  float bv = bias[n0 + q];
#pragma unroll
  for (int r = 0; r < 8; ++r)
    out[(size_t)(m0 + r + 8 * g) * kDim + n0 + q] = acc[r] + bv;
}

// ------------------------------ host side -----------------------------------
struct BlockP {
  const float *ln1_w, *ln1_b, *ln2_w, *ln2_b, *ss1, *ss2, *in_w, *cw, *cb,
      *dt_bias, *A_log, *D, *norm_w, *out_w;
};

struct WS {
  float *t0, *zx, *dtb, *xc, *xdtb, *dAb, *Acs, *states, *prevb, *cdec, *Ybuf,
      *ynb, *mout, *y1, *y2, *y3;
};

void run_block(hipStream_t st, const float* u, float* out, const BlockP& P, const WS& w,
               const float* r1, const float* r2, const float* r3) {
  ln_rows_k<<<kRows / 8, 256, 0, st>>>(u, P.ln1_w, P.ln1_b, w.t0, 1e-6f);
  gemm_wmma_k<<<dim3(kDProj / 16, kRows / 128), 256, 0, st>>>(w.t0, kDim, P.in_w, kDim, w.zx,
                                                              kDProj, kDim);
  dt_k<<<kRows * kNH / 256, 256, 0, st>>>(w.zx, P.dt_bias, w.dtb);
  conv1d_k<<<kRows * kConvDim / 256, 256, 0, st>>>(w.zx, P.cw, P.cb, w.xc);
  xdt_k<<<kRows * kNH / 256, 256, 0, st>>>(w.xc, w.dtb, P.A_log, w.xdtb, w.dAb);
  ssd_chunk_k<<<dim3(kNChunks, kBatch), 256, 0, st>>>(w.dAb, w.xc, w.xdtb, w.Ybuf, w.states,
                                                      w.cdec, w.Acs);
  scan_k<<<kBatch * kNH * kHD * kDState / 256, 256, 0, st>>>(w.states, w.cdec, w.prevb);
  ssd_yoff_k<<<dim3(kNChunks, kBatch), 256, 0, st>>>(w.xc, w.Acs, w.prevb, w.Ybuf);
  gate_rms_k<<<kRows / 8, 256, 0, st>>>(w.Ybuf, w.xc, w.zx, P.D, P.norm_w, w.ynb);
  gemm_wmma_k<<<dim3(kDim / 16, kRows / 128), 256, 0, st>>>(w.ynb, kDInner, P.out_w, kDInner,
                                                            w.mout, kDim, kDInner);
  finalize_k<<<kRows / 8, 256, 0, st>>>(u, w.mout, P.ss1, P.ss2, P.ln2_w, P.ln2_b, r1, r2, r3,
                                        out);
}

}  // namespace

extern "C" void kernel_launch(void* const* d_in, const int* in_sizes, int n_in, void* d_out,
                              int out_size, void* d_ws, size_t ws_size, hipStream_t stream) {
  auto F = [&](int i) { return (const float*)d_in[i]; };
  const float* x = F(0);
  auto mkP = [&](int base) {
    BlockP p;
    p.ln1_w = F(base + 0); p.ln1_b = F(base + 1); p.ln2_w = F(base + 2); p.ln2_b = F(base + 3);
    p.ss1 = F(base + 4); p.ss2 = F(base + 5); p.in_w = F(base + 6); p.cw = F(base + 7);
    p.cb = F(base + 8); p.dt_bias = F(base + 9); p.A_log = F(base + 10); p.D = F(base + 11);
    p.norm_w = F(base + 12); p.out_w = F(base + 13);
    return p;
  };
  BlockP P1 = mkP(1), P2 = mkP(15);
  const float* conv_w = F(29);
  const float* conv_b = F(30);
  const float* pe_w = F(31);
  const float* pe_b = F(32);

  float* ws = (float*)d_ws;
  size_t o = 0;
  auto take = [&](size_t n) { float* p = ws + o; o += n; return p; };
  WS w;
  w.t0 = take((size_t)kRows * kDim);
  w.zx = take((size_t)kRows * kDProj);
  w.dtb = take((size_t)kRows * kNH);
  w.xc = take((size_t)kRows * kConvDim);
  w.xdtb = take((size_t)kRows * kDInner);
  w.dAb = take((size_t)kRows * kNH);
  w.Acs = take((size_t)kBatch * kNChunks * kNH * kChunk);
  w.states = take((size_t)kBatch * kNChunks * kNH * kHD * kDState);
  w.prevb = take((size_t)kBatch * kNChunks * kNH * kHD * kDState);
  w.cdec = take((size_t)kBatch * kNChunks * kNH);
  w.Ybuf = take((size_t)kRows * kDInner);
  w.ynb = take((size_t)kRows * kDInner);
  w.mout = take((size_t)kRows * kDim);
  w.y1 = take((size_t)kRows * kDim);
  w.y2 = take((size_t)kRows * kDim);
  w.y3 = take((size_t)kRows * kDim);

  // y1 = block(x; p1) + x
  run_block(stream, x, w.y1, P1, w, x, nullptr, nullptr);
  // y2 = block(y1; p2) + y1 + x
  run_block(stream, w.y1, w.y2, P2, w, w.y1, x, nullptr);
  // y3 = block(y2; p2) + y2 + y1 + x
  run_block(stream, w.y2, w.y3, P2, w, w.y2, w.y1, x);

  float* out = (float*)d_out;
  conv2d_wmma_k<<<dim3(kDim / 16, kRows / 16), 32, 0, stream>>>(w.y3, conv_w, conv_b, out);
  ln_rows_k<<<kRows / 8, 256, 0, stream>>>(out, pe_w, pe_b, out, 1e-5f);
}
